// FusedMoEBlock_953482740194
// MI455X (gfx1250) — compile-verified, matrix operand order checked by reading, not run on hardware
//
#include <hip/hip_runtime.h>
#include <math.h>

// ---------------------------------------------------------------------------
// Fused MoE (top-2, SwiGLU experts) for gfx1250 / MI455X.
//
// Roofline: routed compute = 283 GFLOP; fp32 weight stream ~553MB. Strategy:
//  1) one-time pass converts x and all expert weights to f16; weights are
//     rewritten in WMMA-B "blob" layout (each 16x32 tile stored as the exact
//     32-lane x 32B register image), so the GEMM hot loop needs NO in-loop
//     transpose, NO LDS staging for operands, and NO per-step barriers.
//  2) grouped expert GEMM on v_wmma_f32_16x16x32_f16 (f32 accumulate),
//     M=64-row tiles; the 64x1408 f16 SwiGLU intermediate lives entirely in
//     CDNA5's 320KB LDS so each block does a full-K reduction and writes its
//     output rows exactly once (no FP atomics => bitwise deterministic).
// ---------------------------------------------------------------------------

typedef _Float16 hv2  __attribute__((ext_vector_type(2)));
typedef _Float16 hv8  __attribute__((ext_vector_type(8)));
typedef _Float16 v16h __attribute__((ext_vector_type(16)));
typedef float    v8f  __attribute__((ext_vector_type(8)));

static constexpr int kH = 2048;   // hidden dim
static constexpr int kI = 1408;   // intermediate dim
static constexpr int kE = 16;     // experts
static constexpr int kMT = 64;    // token-rows per expert tile

// LDS stride for the h buffer (f16 elems), padded so 16 rows x b128 reads
// fall in distinct bank quads.
static constexpr int HSTR = 1416;
static constexpr size_t SMEM_BYTES = (size_t)kMT * HSTR * 2 + kMT * 4;

static __device__ __forceinline__ v16h frag16(const _Float16* lo,
                                              const _Float16* hi) {
  hv8 a = *reinterpret_cast<const hv8*>(lo);
  hv8 b = *reinterpret_cast<const hv8*>(hi);
  return __builtin_shufflevector(a, b, 0, 1, 2, 3, 4, 5, 6, 7, 8, 9, 10, 11,
                                 12, 13, 14, 15);
}

// --------------------------------------------------------------------------
__global__ void moe_zero_counts(int* counts) {
  if (threadIdx.x < kE) counts[threadIdx.x] = 0;
}

// Router: one wave per token; shuffle-reduced logits, stable top-2, renorm.
__global__ void moe_router(const float* __restrict__ x,
                           const float* __restrict__ gate_w,
                           int* __restrict__ counts, int* __restrict__ rowmap,
                           float* __restrict__ topw, int T) {
  const int lane = threadIdx.x & 31;
  const int wid = threadIdx.x >> 5;
  const int t = blockIdx.x * 8 + wid;
  if (t >= T) return;

  float acc[kE];
#pragma unroll
  for (int e = 0; e < kE; ++e) acc[e] = 0.f;

  const float* xrow = x + (size_t)t * kH;
  for (int h = lane; h < kH; h += 32) {
    float xv = xrow[h];
    const float4* g4 = reinterpret_cast<const float4*>(gate_w + h * kE);
    float4 a = g4[0], b = g4[1], c = g4[2], d = g4[3];
    acc[0] += xv * a.x;  acc[1] += xv * a.y;  acc[2] += xv * a.z;  acc[3] += xv * a.w;
    acc[4] += xv * b.x;  acc[5] += xv * b.y;  acc[6] += xv * b.z;  acc[7] += xv * b.w;
    acc[8] += xv * c.x;  acc[9] += xv * c.y;  acc[10] += xv * c.z; acc[11] += xv * c.w;
    acc[12] += xv * d.x; acc[13] += xv * d.y; acc[14] += xv * d.z; acc[15] += xv * d.w;
  }
#pragma unroll
  for (int e = 0; e < kE; ++e) {
#pragma unroll
    for (int off = 16; off > 0; off >>= 1) acc[e] += __shfl_xor(acc[e], off, 32);
  }

  float l0 = acc[0], l1 = -INFINITY;
  int i0 = 0, i1 = 0;
#pragma unroll
  for (int e = 1; e < kE; ++e) {
    float v = acc[e];
    if (v > l0) { l1 = l0; i1 = i0; l0 = v; i0 = e; }
    else if (v > l1) { l1 = v; i1 = e; }
  }
  float d = __expf(l1 - l0);  // <= 1
  float w0 = 1.f / (1.f + d);
  float w1 = d / (1.f + d);

  if (lane == 0) {
    topw[2 * t + 0] = w0;
    topw[2 * t + 1] = w1;
    int p0 = atomicAdd(&counts[i0], 1);
    rowmap[(size_t)i0 * T + p0] = t * 2 + 0;
    int p1 = atomicAdd(&counts[i1], 1);
    rowmap[(size_t)i1 * T + p1] = t * 2 + 1;
  }
}

// fp32 -> f16 cast, 8 elements/thread (b128 in / b128 out)
__global__ void moe_cvt_x(const float* __restrict__ in,
                          _Float16* __restrict__ out, size_t n) {
  size_t i = ((size_t)blockIdx.x * 256 + threadIdx.x) * 8;
  if (i >= n) return;
  float4 a = *reinterpret_cast<const float4*>(in + i);
  float4 b = *reinterpret_cast<const float4*>(in + i + 4);
  hv8 v = {(_Float16)a.x, (_Float16)a.y, (_Float16)a.z, (_Float16)a.w,
           (_Float16)b.x, (_Float16)b.y, (_Float16)b.z, (_Float16)b.w};
  *reinterpret_cast<hv8*>(out + i) = v;
}

// Convert a [K][N] fp32 operand to f16 in WMMA-B blob layout:
// blob(nt,kt) = contiguous 1KB holding the exact 32-lane x 16-f16 register
// image of tile (n in [16nt,16nt+16), k in [32kt,32kt+32)).
//   lane = (n&15) + 16*((k>>3)&1);  elem = (k&7) + 8*((k>>4)&1)
// Grid: (ceil(K/2*N/256), E). Threads handle a k-pair => 4B LDS-free store.
__global__ void moe_cvt_b(const float* __restrict__ src,
                          _Float16* __restrict__ dst, int K, int N) {
  size_t per = (size_t)(K / 2) * N;
  size_t p = (size_t)blockIdx.x * 256 + threadIdx.x;
  if (p >= per) return;
  int n = (int)(p % N);
  int k = (int)(p / N) * 2;
  const float* s = src + (size_t)blockIdx.y * K * N;
  char* d = reinterpret_cast<char*>(dst) + (size_t)blockIdx.y * K * N * 2;
  float a = s[(size_t)k * N + n];
  float b = s[(size_t)(k + 1) * N + n];
  int nt = n >> 4, kt = k >> 5;
  int lane = (n & 15) + (((k >> 3) & 1) << 4);
  int elem = (k & 7) + (((k >> 4) & 1) << 3);  // even
  size_t blob = (size_t)nt * (K >> 5) + kt;
  hv2 v = {(_Float16)a, (_Float16)b};
  *reinterpret_cast<hv2*>(d + blob * 1024 + lane * 32 + elem * 2) = v;
}

// --------------------------------------------------------------------------
// Grouped expert FFN. Block = 512 threads (16 wave32), one (expert, 64-row)
// tile per block. A fragments load straight from f16 x rows; B fragments are
// two coalesced b128 loads from the blob-layout weights. Only 2 barriers.
__global__ __launch_bounds__(512, 1) void moe_expert_ffn(
    const _Float16* __restrict__ xh, const _Float16* __restrict__ w1h,
    const _Float16* __restrict__ w3h, const _Float16* __restrict__ w2h,
    const int* __restrict__ counts, const int* __restrict__ rowmap,
    float* __restrict__ y, int T, int maxTiles) {
  extern __shared__ char smem[];
  _Float16* s_h = reinterpret_cast<_Float16*>(smem);  // [64][HSTR]
  int* s_row = reinterpret_cast<int*>(s_h + (size_t)kMT * HSTR);

  const int e = blockIdx.x / maxTiles;
  const int tile = blockIdx.x - e * maxTiles;
  const int cnt = counts[e];
  const int m0 = tile * kMT;
  if (m0 >= cnt) return;

  const _Float16* w1e = w1h + (size_t)e * kH * kI;
  const _Float16* w3e = w3h + (size_t)e * kH * kI;
  const _Float16* w2e = w2h + (size_t)e * kI * kH;

  const int tid = threadIdx.x;
  const int lane = tid & 31;
  const int wid = tid >> 5;
  const int n_sub = wid & 7;    // 16-col slice of the 128-col tile
  const int m_pair = wid >> 3;  // rows 0-31 or 32-63
  const int lh = lane & 15;
  const int kb2 = (lane >> 4) * 8;  // per-lane K base in f16 A layout

  if (tid < kMT) {
    int gm = m0 + tid;
    s_row[tid] = (gm < cnt) ? rowmap[(size_t)e * T + gm] : -1;
  }
  __syncthreads();

  // per-lane gathered A-row base pointers (invalid rows -> row 0, discarded)
  const int ra0 = s_row[m_pair * 32 + lh];
  const int ra1 = s_row[m_pair * 32 + 16 + lh];
  const _Float16* xr0 = xh + (size_t)(ra0 >= 0 ? (ra0 >> 1) : 0) * kH;
  const _Float16* xr1 = xh + (size_t)(ra1 >= 0 ? (ra1 >> 1) : 0) * kH;

  // ---------------- Step A: g = x@w1, u = x@w3, h = silu(g)*u -------------
  for (int n0 = 0; n0 < kI; n0 += 128) {
    const size_t nt = (size_t)((n0 >> 4) + n_sub);
    const _Float16* b1 = w1e + nt * (kH >> 5) * 512 + lane * 16;
    const _Float16* b3 = w3e + nt * (kH >> 5) * 512 + lane * 16;
    v8f g0 = {}, g1 = {}, u0 = {}, u1 = {};
    for (int kt = 0; kt < (kH >> 5); ++kt) {
      const int k0 = kt << 5;
      v16h a0 = frag16(xr0 + k0 + kb2, xr0 + k0 + 16 + kb2);
      v16h a1 = frag16(xr1 + k0 + kb2, xr1 + k0 + 16 + kb2);
      const _Float16* bp1 = b1 + (size_t)kt * 512;
      const _Float16* bp3 = b3 + (size_t)kt * 512;
      v16h bb1 = frag16(bp1, bp1 + 8);
      v16h bb3 = frag16(bp3, bp3 + 8);
      __builtin_prefetch(bp1 + 512, 0, 0);
      __builtin_prefetch(bp3 + 512, 0, 0);
      g0 = __builtin_amdgcn_wmma_f32_16x16x32_f16(false, a0, false, bb1,
                                                  (short)0, g0, false, false);
      g1 = __builtin_amdgcn_wmma_f32_16x16x32_f16(false, a1, false, bb1,
                                                  (short)0, g1, false, false);
      u0 = __builtin_amdgcn_wmma_f32_16x16x32_f16(false, a0, false, bb3,
                                                  (short)0, u0, false, false);
      u1 = __builtin_amdgcn_wmma_f32_16x16x32_f16(false, a1, false, bb3,
                                                  (short)0, u1, false, false);
    }
    const int col = n0 + n_sub * 16 + lh;     // D: N = lane&15
    const int mb = m_pair * 32 + (lane >> 4) * 8;  // D: M = j + 8*(lane>=16)
#pragma unroll
    for (int j = 0; j < 8; ++j) {
      float g = g0[j], u = u0[j];
      s_h[(mb + j) * HSTR + col] = (_Float16)(g * u / (1.f + __expf(-g)));
      g = g1[j]; u = u1[j];
      s_h[(mb + 16 + j) * HSTR + col] = (_Float16)(g * u / (1.f + __expf(-g)));
    }
  }
  __syncthreads();  // all h columns visible to all waves

  // ---------------- Step B: y = h @ w2 (full-I reduction in-block) --------
  for (int hb = 0; hb < kH; hb += 128) {
    const size_t nt = (size_t)((hb >> 4) + n_sub);
    const _Float16* b2 = w2e + nt * (kI >> 5) * 512 + lane * 16;
    v8f o0 = {}, o1 = {};
    for (int kt = 0; kt < (kI >> 5); ++kt) {
      const int k0 = kt << 5;
      const _Float16* ar = s_h + (m_pair * 32 + lh) * HSTR + k0;
      v16h a0 = frag16(ar + kb2, ar + 16 + kb2);
      const _Float16* ar1 = ar + 16 * HSTR;
      v16h a1 = frag16(ar1 + kb2, ar1 + 16 + kb2);
      const _Float16* bp = b2 + (size_t)kt * 512;
      v16h bb = frag16(bp, bp + 8);
      __builtin_prefetch(bp + 512, 0, 0);
      o0 = __builtin_amdgcn_wmma_f32_16x16x32_f16(false, a0, false, bb,
                                                  (short)0, o0, false, false);
      o1 = __builtin_amdgcn_wmma_f32_16x16x32_f16(false, a1, false, bb,
                                                  (short)0, o1, false, false);
    }
    const int col = hb + n_sub * 16 + lh;
    const int mb = m_pair * 32 + (lane >> 4) * 8;
#pragma unroll
    for (int j = 0; j < 8; ++j) {
      int rid = s_row[mb + j];
      if (rid >= 0) y[(size_t)rid * kH + col] = o0[j];
      rid = s_row[mb + 16 + j];
      if (rid >= 0) y[(size_t)rid * kH + col] = o1[j];
    }
  }
}

// Combine: out[t] = w[t,0]*y[2t] + w[t,1]*y[2t+1] (fixed order, deterministic)
__global__ void moe_combine(const float* __restrict__ y,
                            const float* __restrict__ topw,
                            float* __restrict__ out, int total) {
  int idx = blockIdx.x * 256 + threadIdx.x;
  if (idx >= total) return;
  int t = idx / kH;
  int h = idx - t * kH;
  out[idx] = topw[2 * t] * y[(size_t)(2 * t) * kH + h] +
             topw[2 * t + 1] * y[(size_t)(2 * t + 1) * kH + h];
}

// --------------------------------------------------------------------------
extern "C" void kernel_launch(void* const* d_in, const int* in_sizes, int n_in,
                              void* d_out, int out_size, void* d_ws,
                              size_t ws_size, hipStream_t stream) {
  const float* x = (const float*)d_in[0];       // [T, H]
  const float* gate_w = (const float*)d_in[1];  // [H, E]
  const float* w1 = (const float*)d_in[2];      // [E, H, I]
  const float* w3 = (const float*)d_in[3];      // [E, H, I]
  const float* w2 = (const float*)d_in[4];      // [E, I, H]
  float* out = (float*)d_out;

  const int T = in_sizes[0] / kH;  // 8192

  // workspace layout
  char* ws = (char*)d_ws;
  size_t off = 0;
  auto alloc = [&](size_t bytes) {
    size_t o = off;
    off = (off + bytes + 255) & ~(size_t)255;
    return o;
  };
  int* counts = (int*)(ws + alloc(kE * 4));
  int* rowmap = (int*)(ws + alloc((size_t)kE * T * 4));
  float* topw = (float*)(ws + alloc((size_t)T * 2 * 4));
  _Float16* xh = (_Float16*)(ws + alloc((size_t)T * kH * 2));
  _Float16* w1h = (_Float16*)(ws + alloc((size_t)kE * kH * kI * 2));
  _Float16* w3h = (_Float16*)(ws + alloc((size_t)kE * kH * kI * 2));
  _Float16* w2h = (_Float16*)(ws + alloc((size_t)kE * kI * kH * 2));
  float* y = (float*)(ws + alloc((size_t)T * 2 * kH * 4));

  moe_zero_counts<<<1, 32, 0, stream>>>(counts);
  moe_router<<<(T + 7) / 8, 256, 0, stream>>>(x, gate_w, counts, rowmap, topw, T);

  const size_t nx = (size_t)T * kH;
  moe_cvt_x<<<(unsigned)((nx / 8 + 255) / 256), 256, 0, stream>>>(x, xh, nx);
  {
    const size_t p13 = (size_t)(kH / 2) * kI;  // k-pairs per expert (w1/w3)
    dim3 g13((unsigned)((p13 + 255) / 256), kE);
    moe_cvt_b<<<g13, 256, 0, stream>>>(w1, w1h, kH, kI);
    moe_cvt_b<<<g13, 256, 0, stream>>>(w3, w3h, kH, kI);
    const size_t p2 = (size_t)(kI / 2) * kH;   // k-pairs per expert (w2)
    dim3 g2((unsigned)((p2 + 255) / 256), kE);
    moe_cvt_b<<<g2, 256, 0, stream>>>(w2, w2h, kI, kH);
  }

  const int maxTiles = (T + kMT - 1) / kMT;
  moe_expert_ffn<<<kE * maxTiles, 512, SMEM_BYTES, stream>>>(
      xh, w1h, w3h, w2h, counts, rowmap, y, T, maxTiles);
  moe_combine<<<(unsigned)(((size_t)T * kH + 255) / 256), 256, 0, stream>>>(
      y, topw, out, T * kH);
}
